// NonAttentiveTacatron_78168404787262
// MI455X (gfx1250) — matrix operand order repaired and checked by reading, not compiled, not run on hardware
//
#include <hip/hip_runtime.h>
#include <initializer_list>
#include <math.h>

// ---------------- model dims ----------------
#define BB 32
#define TI 256
#define TO 1024
#define PH 512
#define SPK 128
#define FULL 640        // enc out 512 + spk 128
#define ENCH 256
#define POSD 128
#define NMEL 80
#define DECH 1024
#define RI (TI + 4)     // padded time rows per batch (enc convs)
#define RO (TO + 4)     // padded time rows per batch (postnet)

typedef __attribute__((ext_vector_type(2)))  float  v2f;
typedef __attribute__((ext_vector_type(8)))  float  v8f;
typedef __attribute__((ext_vector_type(8)))  __bf16 v8bf;
typedef __attribute__((ext_vector_type(16))) __bf16 v16bf;

__device__ __forceinline__ float sigf(float x) { return 1.0f / (1.0f + __expf(-x)); }

__device__ __forceinline__ unsigned short f2bf(float f) {
  unsigned int u = __float_as_uint(f);
  u += 0x7FFFu + ((u >> 16) & 1u);            // round-to-nearest-even
  return (unsigned short)(u >> 16);
}

#define GSL(i, n) for (long i = (long)blockIdx.x * 256 + threadIdx.x; i < (long)(n); i += (long)gridDim.x * 256)

// =========================================================================
// f32 multi-operand WMMA GEMM (decoder path): C = act(init + sum_o A_o @ W_o^T)
// =========================================================================
#define MAXOPS 5
struct GOp { const float* A; const float* W; int K; int lda; int ldw; long aB; long wB; };
struct GemmArgs {
  GOp op[MAXOPS]; int nops;
  const float* binit; int bmode; int ldb; long bB;   // 0 none, 1 vector[n], 2 matrix[m][n]
  float* C; int ldc; long cB;
  int M; int N; int act;                             // 0 none, 1 relu, 2 tanh
};

__global__ __launch_bounds__(256) void gemm_multi(GemmArgs g) {
  const int lane = threadIdx.x & 31, wave = threadIdx.x >> 5;
  const int lo = lane & 15, hi = lane >> 4;
  const int mt = wave & 1, nt = wave >> 1;           // 2 x 4 wave layout
  const int m0 = blockIdx.y * 32 + mt * 16;
  const int n0 = blockIdx.x * 64 + nt * 16;
  const int bz = blockIdx.z;
  if (n0 >= g.N || m0 >= g.M) return;                // wave-uniform: EXEC stays all-ones

  v8f acc0, acc1;
  if (g.bmode == 1) {
    float bv = g.binit[(long)bz * g.bB + n0 + lo];
    for (int r = 0; r < 8; ++r) acc0[r] = bv;
  } else if (g.bmode == 2) {
    for (int r = 0; r < 8; ++r) {
      int m = m0 + r + 8 * hi; if (m >= g.M) m = g.M - 1;
      acc0[r] = g.binit[(long)bz * g.bB + (long)m * g.ldb + n0 + lo];
    }
  } else {
    for (int r = 0; r < 8; ++r) acc0[r] = 0.0f;
  }
  for (int r = 0; r < 8; ++r) acc1[r] = 0.0f;

  for (int o = 0; o < g.nops; ++o) {
    GOp p = g.op[o];
    int ma = m0 + lo; if (ma >= g.M) ma = g.M - 1;   // clamp: garbage rows computed, never stored
    const float* Ar = p.A + (long)bz * p.aB + (long)ma * p.lda + 2 * hi;
    const float* Wr = p.W + (long)bz * p.wB + (long)(n0 + lo) * p.ldw + 2 * hi;
    int k = 0;
    for (; k + 8 <= p.K; k += 8) {                   // two independent WMMA chains
      v2f a0 = *(const v2f*)(Ar + k);
      v2f b0 = *(const v2f*)(Wr + k);
      acc0 = __builtin_amdgcn_wmma_f32_16x16x4_f32(false, a0, false, b0, (short)0, acc0, false, false);
      v2f a1 = *(const v2f*)(Ar + k + 4);
      v2f b1 = *(const v2f*)(Wr + k + 4);
      acc1 = __builtin_amdgcn_wmma_f32_16x16x4_f32(false, a1, false, b1, (short)0, acc1, false, false);
    }
    for (; k < p.K; k += 4) {
      v2f a0 = *(const v2f*)(Ar + k);
      v2f b0 = *(const v2f*)(Wr + k);
      acc0 = __builtin_amdgcn_wmma_f32_16x16x4_f32(false, a0, false, b0, (short)0, acc0, false, false);
    }
  }

  for (int r = 0; r < 8; ++r) {
    int m = m0 + r + 8 * hi;
    if (m < g.M) {
      float v = acc0[r] + acc1[r];
      if (g.act == 1) v = fmaxf(v, 0.0f);
      else if (g.act == 2) v = tanhf(v);
      g.C[(long)bz * g.cB + (long)m * g.ldc + n0 + lo] = v;
    }
  }
}

// =========================================================================
// bf16 multi-operand WMMA GEMM (feed-forward bulk): f32 accumulate
// A tile per lane: one contiguous 16-elem chunk (two 8-elem loads); B tile: one contiguous 16-elem chunk
// =========================================================================
struct BOp { const unsigned short* A; const unsigned short* W; int K; int lda; int ldw; long aB; long wB; };
struct BGemmArgs {
  BOp op[MAXOPS]; int nops;
  const float* binit; int bmode; int ldb; long bB;
  float* C; int ldc; long cB;
  int M; int N; int act;
};

__device__ __forceinline__ v16bf load_a_bf(const unsigned short* p) {
  v8bf lo = *(const v8bf*)(p);
  v8bf hi = *(const v8bf*)(p + 8);
  return __builtin_shufflevector(lo, hi, 0, 1, 2, 3, 4, 5, 6, 7, 8, 9, 10, 11, 12, 13, 14, 15);
}

__global__ __launch_bounds__(256) void gemm_bf16(BGemmArgs g) {
  const int lane = threadIdx.x & 31, wave = threadIdx.x >> 5;
  const int lo = lane & 15, hi = lane >> 4;
  const int mt = wave & 1, nt = wave >> 1;
  const int m0 = blockIdx.y * 32 + mt * 16;
  const int n0 = blockIdx.x * 64 + nt * 16;
  const int bz = blockIdx.z;
  if (n0 >= g.N || m0 >= g.M) return;

  v8f acc0, acc1;
  if (g.bmode == 1) {
    float bv = g.binit[(long)bz * g.bB + n0 + lo];
    for (int r = 0; r < 8; ++r) acc0[r] = bv;
  } else if (g.bmode == 2) {
    for (int r = 0; r < 8; ++r) {
      int m = m0 + r + 8 * hi; if (m >= g.M) m = g.M - 1;
      acc0[r] = g.binit[(long)bz * g.bB + (long)m * g.ldb + n0 + lo];
    }
  } else {
    for (int r = 0; r < 8; ++r) acc0[r] = 0.0f;
  }
  for (int r = 0; r < 8; ++r) acc1[r] = 0.0f;

  for (int o = 0; o < g.nops; ++o) {
    BOp p = g.op[o];
    int ma = m0 + lo; if (ma >= g.M) ma = g.M - 1;
    const unsigned short* Ar = p.A + (long)bz * p.aB + (long)ma * p.lda + 16 * hi;
    const unsigned short* Wr = p.W + (long)bz * p.wB + (long)(n0 + lo) * p.ldw + 16 * hi;
    int k = 0;
    for (; k + 64 <= p.K; k += 64) {
      v16bf a0 = load_a_bf(Ar + k);
      v16bf b0 = *(const v16bf*)(Wr + k);
      acc0 = __builtin_amdgcn_wmma_f32_16x16x32_bf16(false, a0, false, b0, (short)0, acc0, false, false);
      v16bf a1 = load_a_bf(Ar + k + 32);
      v16bf b1 = *(const v16bf*)(Wr + k + 32);
      acc1 = __builtin_amdgcn_wmma_f32_16x16x32_bf16(false, a1, false, b1, (short)0, acc1, false, false);
    }
    for (; k < p.K; k += 32) {
      v16bf a0 = load_a_bf(Ar + k);
      v16bf b0 = *(const v16bf*)(Wr + k);
      acc0 = __builtin_amdgcn_wmma_f32_16x16x32_bf16(false, a0, false, b0, (short)0, acc0, false, false);
    }
  }

  for (int r = 0; r < 8; ++r) {
    int m = m0 + r + 8 * hi;
    if (m < g.M) {
      float v = acc0[r] + acc1[r];
      if (g.act == 1) v = fmaxf(v, 0.0f);
      else if (g.act == 2) v = tanhf(v);
      g.C[(long)bz * g.cB + (long)m * g.ldc + n0 + lo] = v;
    }
  }
}

// =========================================================================
// fused LSTM cell step (f32 gate GEMM + LDS staging + nonlinearity), M == 32
// =========================================================================
struct CellArgs {
  GOp op[4]; int nops;
  const float* binit; int bmode; int ldb;            // 1 vector[4H], 2 matrix[m][4H]
  float* h_out; float* c_st; float* hseq; long ldseq; int H;
};

__global__ __launch_bounds__(256) void lstm_cell(CellArgs a) {
  __shared__ float gb[4][32][64];
  const int lane = threadIdx.x & 31, wave = threadIdx.x >> 5;
  const int lo = lane & 15, hi = lane >> 4;
  const int gate = wave & 3, mt = wave >> 2;         // 4 gates x 2 M-halves
  const int hs0 = blockIdx.x * 64;                   // H-slice of 64
  const int m0 = mt * 16;

  v8f acc[4];
  for (int ntile = 0; ntile < 4; ++ntile) {
    int n = gate * a.H + hs0 + ntile * 16 + lo;
    if (a.bmode == 1) {
      float bv = a.binit[n];
      for (int r = 0; r < 8; ++r) acc[ntile][r] = bv;
    } else {
      for (int r = 0; r < 8; ++r) {
        int m = m0 + r + 8 * hi;
        acc[ntile][r] = a.binit[(long)m * a.ldb + n];
      }
    }
  }
  for (int o = 0; o < a.nops; ++o) {
    GOp p = a.op[o];
    const float* Ar = p.A + (long)(m0 + lo) * p.lda + 2 * hi;
    const float* W0 = p.W + 2 * hi;
    for (int k = 0; k < p.K; k += 4) {
      v2f av = *(const v2f*)(Ar + k);
      for (int ntile = 0; ntile < 4; ++ntile) {
        int n = gate * a.H + hs0 + ntile * 16 + lo;
        v2f bv = *(const v2f*)(W0 + (long)n * p.ldw + k);
        acc[ntile] = __builtin_amdgcn_wmma_f32_16x16x4_f32(false, av, false, bv, (short)0, acc[ntile], false, false);
      }
    }
  }
  for (int ntile = 0; ntile < 4; ++ntile)
    for (int r = 0; r < 8; ++r)
      gb[gate][m0 + r + 8 * hi][ntile * 16 + lo] = acc[ntile][r];
  __syncthreads();
  for (int e = threadIdx.x; e < 32 * 64; e += 256) {
    int m = e >> 6, hh = e & 63;
    int col = hs0 + hh;
    float gi = gb[0][m][hh], gf = gb[1][m][hh], gg = gb[2][m][hh], go = gb[3][m][hh];
    float cp = a.c_st[m * a.H + col];
    float cn = sigf(gf) * cp + sigf(gi) * tanhf(gg);
    float hn = sigf(go) * tanhf(cn);
    a.c_st[m * a.H + col] = cn;
    a.h_out[m * a.H + col] = hn;
    if (a.hseq) a.hseq[(long)m * a.ldseq + col] = hn;
  }
}

// ---------------- elementwise / helper kernels ----------------
__global__ void fill_f32(float* p, long n, float v) { GSL(i, n) p[i] = v; }
__global__ void fill_u16(unsigned short* p, long n) { GSL(i, n) p[i] = 0; }
__global__ void vadd_k(float* o, const float* x, const float* y, long n) { GSL(i, n) o[i] = x[i] + y[i]; }
__global__ void cvt_flat_bf(const float* s, unsigned short* d, long n) { GSL(i, n) d[i] = f2bf(s[i]); }

__global__ void cvt_rows_bf(const float* src, int ldin, unsigned short* dst, int ldout, long rows, int K) {
  GSL(i, rows * (long)K) {
    long r = i / K; int k = (int)(i % K);
    dst[r * (long)ldout + k] = f2bf(src[r * (long)ldin + k]);
  }
}

__global__ void embed_ph(const int* ph, const float* tab, float* X) {
  GSL(i, (long)BB * TI * PH) {
    int c = (int)(i % PH); long bt = i / PH; int b = (int)(bt / TI), t = (int)(bt % TI);
    X[((long)b * RI + 2 + t) * PH + c] = tab[(long)ph[bt] * PH + c];
  }
}

// conv weight (Cout,Cin,5) -> 5 bf16 matrices [k][co][Kp], K zero-padded to Kp
__global__ void repack_conv_bf(const float* W, unsigned short* wk, int Cout, int Cin, int Kp) {
  GSL(i, (long)5 * Cout * Kp) {
    int ci = (int)(i % Kp); long r = i / Kp; int co = (int)(r % Cout); int k = (int)(r / Cout);
    float v = (ci < Cin) ? W[((long)co * Cin + ci) * 5 + k] : 0.0f;
    wk[i] = f2bf(v);
  }
}

__global__ void bn_stats(const float* Y, int C, int Bn, int T, int R, float* mean, float* istd) {
  int c = blockIdx.x;
  __shared__ float s1[256], s2[256];
  float a = 0.0f, b = 0.0f;
  for (int r = threadIdx.x; r < Bn * T; r += 256) {
    int bb = r / T, t = r % T;
    float v = Y[((long)bb * R + t) * C + c];
    a += v; b += v * v;
  }
  s1[threadIdx.x] = a; s2[threadIdx.x] = b; __syncthreads();
  for (int s = 128; s > 0; s >>= 1) {
    if (threadIdx.x < s) { s1[threadIdx.x] += s1[threadIdx.x + s]; s2[threadIdx.x] += s2[threadIdx.x + s]; }
    __syncthreads();
  }
  if (threadIdx.x == 0) {
    float m = s1[0] / (float)(Bn * T);
    float v = s2[0] / (float)(Bn * T) - m * m;
    mean[c] = m; istd[c] = rsqrtf(v + 1e-5f);
  }
}

__global__ void bn_apply(const float* Y, float* X, const float* mean, const float* istd,
                         const float* ga, const float* be, int C, int Bn, int T, int R, int act) {
  GSL(i, (long)Bn * T * C) {
    int c = (int)(i % C); long bt = i / C; int b = (int)(bt / T), t = (int)(bt % T);
    float v = Y[((long)b * R + t) * C + c];
    v = ga[c] * (v - mean[c]) * istd[c] + be[c];
    if (act == 1) v = fmaxf(v, 0.0f);
    else if (act == 2) v = tanhf(v);
    X[((long)b * R + t + 2) * C + c] = v;
  }
}

__global__ void bn_final(const float* Y, const float* mean, const float* istd,
                         const float* ga, const float* be, const float* melo, float* melp) {
  GSL(i, (long)BB * TO * NMEL) {
    int c = (int)(i % NMEL); long bt = i / NMEL; int b = (int)(bt / TO), t = (int)(bt % TO);
    float v = Y[((long)b * RO + t) * NMEL + c];
    melp[i] = melo[i] + ga[c] * (v - mean[c]) * istd[c] + be[c];
  }
}

__global__ void concat_spk(const float* ench, const float* tab, const int* spk, float* emb) {
  GSL(i, (long)BB * TI * FULL) {
    int c = (int)(i % FULL); long bt = i / FULL; int b = (int)(bt / TI);
    emb[i] = (c < PH) ? ench[bt * PH + c] : tab[(long)spk[b] * SPK + (c - PH)];
  }
}

__global__ void rowdot(const float* A, int K, int lda, const float* w, const float* bias, float* out, long n) {
  GSL(r, n) {
    float s = bias[0];
    for (int k = 0; k < K; ++k) s += A[(long)r * lda + k] * w[k];
    out[r] = s;
  }
}

__global__ void fold_dur(float* xw, const float* dur, const float* Wih) {
  GSL(i, (long)BB * TI * 1024) {
    int nn = (int)(i % 1024); long r = i / 1024;
    xw[i] += dur[r] * Wih[(long)nn * 641 + 640];
  }
}

__global__ void centers_k(const float* yd, float* cent) {
  GSL(i, (long)BB * TI) {
    int b = (int)(i / TI), ii = (int)(i % TI);
    float s = 0.0f;
    for (int j = 0; j <= ii; ++j) s += yd[b * TI + j];
    cent[i] = s - 0.5f * yd[i];
  }
}

__global__ void wgauss(const float* cent, const float* rngv, float* wt) {
  int blk = blockIdx.x; int b = blk >> 2; int t = ((blk & 3) << 8) + threadIdx.x;
  __shared__ float cs[TI], is[TI];
  { int i = threadIdx.x; cs[i] = cent[b * TI + i]; float r = rngv[b * TI + i]; is[i] = 1.0f / (r * r); }
  __syncthreads();
  float s = 0.0f; float tf = (float)t;
  long base = ((long)b * TO + t) * TI;
  for (int i = 0; i < TI; ++i) { float d = tf - cs[i]; float e = __expf(-is[i] * d * d); wt[base + i] = e; s += e; }
  float sc = 1.0f / (s + 1e-6f);
  for (int i = 0; i < TI; ++i) wt[base + i] *= sc;
}

__global__ void transp_emb_bf(const float* emb, unsigned short* eT) {
  GSL(i, (long)BB * FULL * TI) {
    int ii = (int)(i % TI); long r = i / TI; int d = (int)(r % FULL); int b = (int)(r / FULL);
    eT[i] = f2bf(emb[((long)b * TI + ii) * FULL + d]);
  }
}

__global__ void pe_k(float* pe) {
  GSL(i, (long)TO * POSD) {
    int d = (int)(i % POSD); int t = (int)(i / POSD); int j = d >> 1;
    float f = __expf(-(float)(2 * j) * logf(10000.0f) / (float)POSD);
    pe[i] = (d & 1) ? cosf((float)t * f) : sinf((float)t * f);
  }
}

__global__ void melcopy(const float* melo, float* X) {
  GSL(i, (long)BB * TO * NMEL) {
    int c = (int)(i % NMEL); long bt = i / NMEL; int b = (int)(bt / TO), t = (int)(bt % TO);
    X[((long)b * RO + t + 2) * NMEL + c] = melo[i];
  }
}

// ---------------- host orchestration ----------------
static inline unsigned nblk(long n) { long b = (n + 255) / 256; return (unsigned)(b > 8192 ? 8192 : b); }

extern "C" void kernel_launch(void* const* d_in, const int* in_sizes, int n_in,
                              void* d_out, int out_size, void* d_ws, size_t ws_size,
                              hipStream_t stream) {
  (void)in_sizes; (void)n_in; (void)out_size; (void)ws_size;
  // input order: setup_inputs() insertion order, recursive over nested dicts/lists
  const int* phon = (const int*)d_in[0];
  const int* spks = (const int*)d_in[1];
  const float* ydur = (const float*)d_in[3];
  const float* ph_tab = (const float*)d_in[4];
  const float* spk_tab = (const float*)d_in[5];
  auto F = [&](int i) { return (const float*)d_in[i]; };
  // 6..17 enc_convs (W,b,gamma,beta)x3 | 18..25 enc_lstm fwd/bwd (Wih,Whh,bih,bhh)
  // 26..33 dur_lstm | 34,35 dur_proj | 36..43 rng_lstm | 44,45 rng_proj
  // 46..49 dec_l0 | 50..53 dec_l1 | 54,55 dec_proj | 56..75 post_convs x5

  float* ws = (float*)d_ws;
  size_t off = 0;
  auto alloc = [&](size_t n) { float* p = ws + off; off += (n + 63) & ~(size_t)63; return p; };
  auto allocU = [&](size_t n) { return (unsigned short*)alloc((n + 1) / 2); };
  float* pad_a = alloc((size_t)BB * RI * PH + 4 * PH);
  float* pad_b = alloc((size_t)BB * RI * PH + 4 * PH);
  float* xw    = alloc((size_t)BB * RI * 1024);
  float* ench  = alloc((size_t)BB * TI * PH);
  float* emb   = alloc((size_t)BB * TI * FULL);
  float* bh    = alloc((size_t)BB * TI * 512);
  float* cent  = alloc(BB * TI);
  float* rngv  = alloc(BB * TI);
  float* wt    = alloc((size_t)BB * TO * TI);
  float* att   = alloc((size_t)BB * TO * FULL);
  float* pe    = alloc((size_t)TO * POSD);
  float* stats = alloc(2 * 512);
  float* bs    = alloc(8 * 4096);
  float* st    = alloc((size_t)12 * 32 * 1024);
  float* zb    = alloc(128);
  float* postA = alloc((size_t)BB * RO * 512 + 4 * 512);
  float* postB = alloc((size_t)BB * RO * 512 + 4 * 512);
  unsigned short* wkbf   = allocU((size_t)5 * 512 * 512);
  unsigned short* wihbf  = allocU((size_t)1024 * 640);
  unsigned short* pada_bf = allocU((size_t)BB * RI * PH + 4 * PH);
  unsigned short* emb_bf  = allocU((size_t)BB * TI * FULL);
  unsigned short* wt_bf   = allocU((size_t)BB * TO * TI);
  unsigned short* embT_bf = allocU((size_t)BB * FULL * TI);
  unsigned short* post_bf = allocU((size_t)BB * RO * 512 + 4 * 512);

  float* durs = (float*)d_out;                       // (32,256,1)
  float* melo = (float*)d_out + BB * TI;             // (32,1024,80)
  float* melp = melo + (size_t)BB * TO * NMEL;       // (32,1024,80)

  auto fill = [&](float* p, long n) { fill_f32<<<nblk(n), 256, 0, stream>>>(p, n, 0.0f); };
  auto vadd = [&](float* o, const float* a, const float* b, long n) {
    vadd_k<<<nblk(n), 256, 0, stream>>>(o, a, b, n);
  };
  auto gemm = [&](std::initializer_list<GOp> ops, const float* binit, int bmode, int ldb, long bB,
                  float* C, int ldc, long cB, int M, int N, int act, int batches) {
    GemmArgs g{}; int i = 0;
    for (auto& o : ops) g.op[i++] = o;
    g.nops = i; g.binit = binit; g.bmode = bmode; g.ldb = ldb; g.bB = bB;
    g.C = C; g.ldc = ldc; g.cB = cB; g.M = M; g.N = N; g.act = act;
    dim3 gr((N + 63) / 64, (M + 31) / 32, batches);
    gemm_multi<<<gr, 256, 0, stream>>>(g);
  };
  auto bgemm = [&](std::initializer_list<BOp> ops, const float* binit, int bmode, int ldb, long bB,
                   float* C, int ldc, long cB, int M, int N, int act, int batches) {
    BGemmArgs g{}; int i = 0;
    for (auto& o : ops) g.op[i++] = o;
    g.nops = i; g.binit = binit; g.bmode = bmode; g.ldb = ldb; g.bB = bB;
    g.C = C; g.ldc = ldc; g.cB = cB; g.M = M; g.N = N; g.act = act;
    dim3 gr((N + 63) / 64, (M + 31) / 32, batches);
    gemm_bf16<<<gr, 256, 0, stream>>>(g);
  };
  auto cell = [&](std::initializer_list<GOp> ops, const float* binit, int bmode, int ldb,
                  float* hout, float* cst, float* hseq, long ldseq, int H) {
    CellArgs a{}; int i = 0;
    for (auto& o : ops) a.op[i++] = o;
    a.nops = i; a.binit = binit; a.bmode = bmode; a.ldb = ldb;
    a.h_out = hout; a.c_st = cst; a.hseq = hseq; a.ldseq = ldseq; a.H = H;
    lstm_cell<<<dim3(H / 64), 256, 0, stream>>>(a);
  };

  // ---- init scratch read-before-write ----
  fill(pad_a, (long)BB * RI * PH + 4 * PH);
  fill(pad_b, (long)BB * RI * PH + 4 * PH);
  fill(postA, (long)BB * RO * 512 + 4 * 512);
  fill(postB, (long)BB * RO * 512 + 4 * 512);
  fill(zb, 128);

  // ---- bias sums (bih + bhh) ----
  vadd(bs + 0 * 4096, F(20), F(21), 1024); // enc fwd
  vadd(bs + 1 * 4096, F(24), F(25), 1024); // enc bwd
  vadd(bs + 2 * 4096, F(28), F(29), 1024); // dur fwd
  vadd(bs + 3 * 4096, F(32), F(33), 1024); // dur bwd
  vadd(bs + 4 * 4096, F(38), F(39), 1024); // rng fwd
  vadd(bs + 5 * 4096, F(42), F(43), 1024); // rng bwd
  vadd(bs + 6 * 4096, F(48), F(49), 4096); // dec_l0
  vadd(bs + 7 * 4096, F(52), F(53), 4096); // dec_l1

  // ---- embedding into padded layout ----
  embed_ph<<<nblk((long)BB * TI * PH), 256, 0, stream>>>(phon, ph_tab, pad_a);

  // ---- encoder convs: 3 x (conv5 + bias + BN + relu), bf16 WMMA ----
  for (int l = 0; l < 3; ++l) {
    repack_conv_bf<<<nblk((long)5 * 512 * 512), 256, 0, stream>>>(F(6 + 4 * l), wkbf, 512, 512, 512);
    cvt_flat_bf<<<nblk((long)BB * RI * PH + 4 * PH), 256, 0, stream>>>(pad_a, pada_bf, (long)BB * RI * PH + 4 * PH);
    bgemm({ BOp{pada_bf + 0 * PH, wkbf + 0L * 512 * 512, 512, 512, 512, 0, 0},
            BOp{pada_bf + 1 * PH, wkbf + 1L * 512 * 512, 512, 512, 512, 0, 0},
            BOp{pada_bf + 2 * PH, wkbf + 2L * 512 * 512, 512, 512, 512, 0, 0},
            BOp{pada_bf + 3 * PH, wkbf + 3L * 512 * 512, 512, 512, 512, 0, 0},
            BOp{pada_bf + 4 * PH, wkbf + 4L * 512 * 512, 512, 512, 512, 0, 0} },
          F(7 + 4 * l), 1, 0, 0, pad_b, 512, 0, BB * RI, 512, 0, 1);
    bn_stats<<<512, 256, 0, stream>>>(pad_b, 512, BB, TI, RI, stats, stats + 512);
    bn_apply<<<nblk((long)BB * TI * 512), 256, 0, stream>>>(pad_b, pad_a, stats, stats + 512,
                                                            F(8 + 4 * l), F(9 + 4 * l), 512, BB, TI, RI, 1);
  }
  cvt_flat_bf<<<nblk((long)BB * RI * PH + 4 * PH), 256, 0, stream>>>(pad_a, pada_bf, (long)BB * RI * PH + 4 * PH);

  // ---- encoder BiLSTM (H=256) ----
  float* h0s = st + 0 * 32768; float* h1s = st + 1 * 32768; float* c0s = st + 2 * 32768;
  for (int dir = 0; dir < 2; ++dir) {
    const float* Whh = F(19 + 4 * dir);
    cvt_rows_bf<<<nblk((long)1024 * 512), 256, 0, stream>>>(F(18 + 4 * dir), 512, wihbf, 512, 1024, 512);
    bgemm({ BOp{pada_bf, wihbf, 512, 512, 512, 0, 0} },
          bs + dir * 4096, 1, 0, 0, xw, 1024, 0, BB * RI, 1024, 0, 1);
    fill(h0s, 32 * ENCH); fill(c0s, 32 * ENCH);
    for (int s = 0; s < TI; ++s) {
      int t = dir ? (TI - 1 - s) : s;
      float* hin = (s & 1) ? h1s : h0s;
      float* hout = (s & 1) ? h0s : h1s;
      cell({ GOp{hin, Whh, ENCH, ENCH, ENCH, 0, 0} },
           xw + (size_t)(2 + t) * 1024, 2, RI * 1024,
           hout, c0s, ench + (size_t)t * PH + dir * ENCH, (long)TI * PH, ENCH);
    }
  }

  // ---- emb = concat(ench, spk) ----
  concat_spk<<<nblk((long)BB * TI * FULL), 256, 0, stream>>>(ench, spk_tab, spks, emb);
  cvt_flat_bf<<<nblk((long)BB * TI * FULL), 256, 0, stream>>>(emb, emb_bf, (long)BB * TI * FULL);

  // ---- duration BiLSTM + projection ----
  for (int dir = 0; dir < 2; ++dir) {
    const float* Whh = F(27 + 4 * dir);
    cvt_rows_bf<<<nblk((long)1024 * 640), 256, 0, stream>>>(F(26 + 4 * dir), 640, wihbf, 640, 1024, 640);
    bgemm({ BOp{emb_bf, wihbf, FULL, FULL, FULL, 0, 0} },
          bs + (2 + dir) * 4096, 1, 0, 0, xw, 1024, 0, BB * TI, 1024, 0, 1);
    fill(h0s, 32 * ENCH); fill(c0s, 32 * ENCH);
    for (int s = 0; s < TI; ++s) {
      int t = dir ? (TI - 1 - s) : s;
      float* hin = (s & 1) ? h1s : h0s;
      float* hout = (s & 1) ? h0s : h1s;
      cell({ GOp{hin, Whh, ENCH, ENCH, ENCH, 0, 0} },
           xw + (size_t)t * 1024, 2, TI * 1024,
           hout, c0s, bh + (size_t)t * 512 + dir * ENCH, (long)TI * 512, ENCH);
    }
  }
  rowdot<<<nblk(BB * TI), 256, 0, stream>>>(bh, 512, 512, F(34), F(35), durs, BB * TI);

  // ---- range BiLSTM + projection (input = [emb, durations]) ----
  for (int dir = 0; dir < 2; ++dir) {
    const float* Wih = F(36 + 4 * dir);
    const float* Whh = F(37 + 4 * dir);
    cvt_rows_bf<<<nblk((long)1024 * 640), 256, 0, stream>>>(Wih, 641, wihbf, 640, 1024, 640);
    bgemm({ BOp{emb_bf, wihbf, FULL, FULL, FULL, 0, 0} },
          bs + (4 + dir) * 4096, 1, 0, 0, xw, 1024, 0, BB * TI, 1024, 0, 1);
    fold_dur<<<nblk((long)BB * TI * 1024), 256, 0, stream>>>(xw, durs, Wih);
    fill(h0s, 32 * ENCH); fill(c0s, 32 * ENCH);
    for (int s = 0; s < TI; ++s) {
      int t = dir ? (TI - 1 - s) : s;
      float* hin = (s & 1) ? h1s : h0s;
      float* hout = (s & 1) ? h0s : h1s;
      cell({ GOp{hin, Whh, ENCH, ENCH, ENCH, 0, 0} },
           xw + (size_t)t * 1024, 2, TI * 1024,
           hout, c0s, bh + (size_t)t * 512 + dir * ENCH, (long)TI * 512, ENCH);
    }
  }
  rowdot<<<nblk(BB * TI), 256, 0, stream>>>(bh, 512, 512, F(44), F(45), rngv, BB * TI);

  // ---- Gaussian upsampling attention (bf16 batched GEMM) ----
  centers_k<<<nblk(BB * TI), 256, 0, stream>>>(ydur, cent);
  wgauss<<<BB * 4, 256, 0, stream>>>(cent, rngv, wt);
  cvt_flat_bf<<<nblk((long)BB * TO * TI), 256, 0, stream>>>(wt, wt_bf, (long)BB * TO * TI);
  transp_emb_bf<<<nblk((long)BB * FULL * TI), 256, 0, stream>>>(emb, embT_bf);
  bgemm({ BOp{wt_bf, embT_bf, TI, TI, TI, (long)TO * TI, (long)FULL * TI} },
        nullptr, 0, 0, 0, att, FULL, (long)TO * FULL, TO, FULL, 0, BB);
  pe_k<<<nblk((long)TO * POSD), 256, 0, stream>>>(pe);

  // ---- decoder: 1024 sequential steps, f32 WMMA (latency-bound recurrence) ----
  const float* Wih0 = F(46); const float* Whh0 = F(47);
  const float* Wih1 = F(50); const float* Whh1 = F(51);
  const float* Wpr = F(54);  const float* bpr = F(55);
  float* H0[2] = { st + 3 * 32768, st + 4 * 32768 };
  float* C0 = st + 5 * 32768;
  float* H1[2] = { st + 6 * 32768, st + 7 * 32768 };
  float* C1 = st + 8 * 32768;
  fill(H0[0], 32 * DECH); fill(C0, 32 * DECH);
  fill(H1[0], 32 * DECH); fill(C1, 32 * DECH);
  for (int t = 0; t < TO; ++t) {
    float* h0in = H0[t & 1]; float* h0out = H0[1 - (t & 1)];
    float* h1in = H1[t & 1]; float* h1out = H1[1 - (t & 1)];
    const float* prev = (t == 0) ? zb : (melo + (size_t)(t - 1) * NMEL);
    int prevlda = (t == 0) ? 0 : TO * NMEL;
    cell({ GOp{prev, Wih0, NMEL, prevlda, 848, 0, 0},
           GOp{att + (size_t)t * FULL, Wih0 + NMEL, FULL, TO * FULL, 848, 0, 0},
           GOp{pe + (size_t)t * POSD, Wih0 + NMEL + FULL, POSD, 0, 848, 0, 0},
           GOp{h0in, Whh0, DECH, DECH, DECH, 0, 0} },
         bs + 6 * 4096, 1, 0, h0out, C0, nullptr, 0, DECH);
    cell({ GOp{h0out, Wih1, DECH, DECH, DECH, 0, 0},
           GOp{h1in, Whh1, DECH, DECH, DECH, 0, 0} },
         bs + 7 * 4096, 1, 0, h1out, C1, nullptr, 0, DECH);
    gemm({ GOp{h1out, Wpr, DECH, DECH, 1792, 0, 0},
           GOp{att + (size_t)t * FULL, Wpr + DECH, FULL, TO * FULL, 1792, 0, 0},
           GOp{pe + (size_t)t * POSD, Wpr + DECH + FULL, POSD, 0, 1792, 0, 0} },
         bpr, 1, 0, 0, melo + (size_t)t * NMEL, TO * NMEL, 0, 32, NMEL, 0, 1);
  }

  // ---- postnet: 5 convs, bf16 WMMA (Cin=80 zero-padded to K=96) ----
  melcopy<<<nblk((long)BB * TO * NMEL), 256, 0, stream>>>(melo, postA);
  for (int l = 0; l < 5; ++l) {
    int Cin = (l == 0) ? NMEL : 512;
    int Cout = (l == 4) ? NMEL : 512;
    int Kp = (l == 0) ? 96 : 512;
    repack_conv_bf<<<nblk((long)5 * Cout * Kp), 256, 0, stream>>>(F(56 + 4 * l), wkbf, Cout, Cin, Kp);
    if (l == 0) {
      fill_u16<<<nblk((long)BB * RO * 96 + 4 * 96), 256, 0, stream>>>(post_bf, (long)BB * RO * 96 + 4 * 96);
      cvt_rows_bf<<<nblk(((long)BB * RO + 4) * NMEL), 256, 0, stream>>>(postA, NMEL, post_bf, 96,
                                                                        (long)BB * RO + 4, NMEL);
    } else {
      cvt_flat_bf<<<nblk((long)BB * RO * 512 + 4 * 512), 256, 0, stream>>>(postA, post_bf,
                                                                           (long)BB * RO * 512 + 4 * 512);
    }
    bgemm({ BOp{post_bf + 0 * Kp, wkbf + 0L * Cout * Kp, Kp, Kp, Kp, 0, 0},
            BOp{post_bf + 1 * Kp, wkbf + 1L * Cout * Kp, Kp, Kp, Kp, 0, 0},
            BOp{post_bf + 2 * Kp, wkbf + 2L * Cout * Kp, Kp, Kp, Kp, 0, 0},
            BOp{post_bf + 3 * Kp, wkbf + 3L * Cout * Kp, Kp, Kp, Kp, 0, 0},
            BOp{post_bf + 4 * Kp, wkbf + 4L * Cout * Kp, Kp, Kp, Kp, 0, 0} },
          F(57 + 4 * l), 1, 0, 0, postB, Cout, 0, BB * RO, Cout, 0, 1);
    bn_stats<<<Cout, 256, 0, stream>>>(postB, Cout, BB, TO, RO, stats, stats + 512);
    if (l < 4) {
      if (l == 0) fill(postA, (long)BB * RO * 512 + 4 * 512);  // clear stale 80-ch layout
      bn_apply<<<nblk((long)BB * TO * Cout), 256, 0, stream>>>(postB, postA, stats, stats + 512,
                                                               F(58 + 4 * l), F(59 + 4 * l),
                                                               Cout, BB, TO, RO, 2);
    } else {
      bn_final<<<nblk((long)BB * TO * NMEL), 256, 0, stream>>>(postB, stats, stats + 512,
                                                               F(58 + 4 * l), F(59 + 4 * l), melo, melp);
    }
  }
}
